// MLA_26422638805114
// MI455X (gfx1250) — matrix operand order repaired and checked by reading, not compiled, not run on hardware
//
#include <hip/hip_runtime.h>

// ---------------------------------------------------------------------------
// MLA forward for MI455X (gfx1250, wave32, WMMA bf16 -> f32 accumulate)
// Async global->LDS tile DMA (ASYNCcnt) + double-buffered LDS pipeline.
// B=4 S=2048 D=2048 H=16 NOPE=128 ROPE=64 VD=128 QK=192 KVR=512
// ---------------------------------------------------------------------------

typedef __attribute__((ext_vector_type(16))) __bf16 v16bf;
typedef __attribute__((ext_vector_type(8)))  __bf16 v8bf;
typedef __attribute__((ext_vector_type(8)))  float  v8f;
typedef unsigned long long u64;

union V16U { v16bf v; v8bf h[2]; };

#define WMMA_BF16(A, Bm, C) \
  __builtin_amdgcn_wmma_f32_16x16x32_bf16(false, (A), false, (Bm), (short)0, (C), false, false)

#define LDSTR 40     // 32 + 8 pad (bf16 elems) -> 80B row stride, conflict free
#define KSTR  200    // 192 + 8 pad for K tiles in attention

#define WAIT_ASYNC_0 asm volatile("s_wait_asynccnt 0x0" ::: "memory")
#define WAIT_ASYNC_4 asm volatile("s_wait_asynccnt 0x4" ::: "memory")
#define WAIT_ASYNC_5 asm volatile("s_wait_asynccnt 0x5" ::: "memory")

static constexpr int CB   = 4;
static constexpr int CS   = 2048;
static constexpr int CD   = 2048;
static constexpr int CH   = 16;
static constexpr int CQK  = 192;
static constexpr int CVD  = 128;
static constexpr int CKVR = 512;
static constexpr int CE1  = CH * CQK + CKVR + 64;   // 3648
static constexpr int CM   = CB * CS;                // 8192

// low 32 bits of a generic pointer to __shared__ = workgroup-relative LDS byte
// offset (flat aperture)
__device__ inline unsigned lds_addr32(const void* p) {
  return (unsigned)(uintptr_t)p;
}

// 16B global -> LDS async copy (per lane). VDST = LDS byte addr, VADDR = 64b.
__device__ inline void async_copy16(unsigned lds_byte, u64 gaddr) {
  asm volatile("global_load_async_to_lds_b128 %0, %1, off"
               :: "v"(lds_byte), "v"(gaddr) : "memory");
}

// ------------------------- fp32 -> bf16 convert ----------------------------
__global__ __launch_bounds__(256) void cvt_kernel(const float* __restrict__ in,
                                                  __bf16* __restrict__ out, int n) {
  int i = blockIdx.x * blockDim.x + threadIdx.x;
  int stride = gridDim.x * blockDim.x;
  for (; i < n; i += stride) out[i] = (__bf16)in[i];
}

// ------------------------- fragment loaders (ISA layouts) ------------------
// A 16x32 bf16: lane<16 -> K {0..7,16..23}; lane>=16 -> K {8..15,24..31}; M = lane&15
__device__ inline v16bf frag_a_lds(const __bf16* lds, int row_base, int lane) {
  int row = row_base + (lane & 15);
  int kb  = (lane < 16) ? 0 : 8;
  V16U u;
  u.h[0] = *(const v8bf*)(lds + row * LDSTR + kb);
  u.h[1] = *(const v8bf*)(lds + row * LDSTR + kb + 16);
  return u.v;
}
// B 32x16 bf16: lane<16 -> K {0..7,16..23}; lane>=16 -> K {8..15,24..31}; N = lane&15
__device__ inline v16bf frag_b_lds(const __bf16* lds, int col_base, int lane) {
  int col = col_base + (lane & 15);
  int kb  = (lane < 16) ? 0 : 8;
  V16U u;
  u.h[0] = *(const v8bf*)(lds + col * LDSTR + kb);
  u.h[1] = *(const v8bf*)(lds + col * LDSTR + kb + 16);
  return u.v;
}

// ------------------------- GEMM tile DMA -----------------------------------
// 4 async 16B copies per thread: A tile 128x32 + B tile 128x32 (N-clamped).
__device__ inline void gemm_issue(const __bf16* __restrict__ A,
                                  const __bf16* __restrict__ Bm,
                                  int N, int K, int m0, int n0, int k0,
                                  unsigned aoff, unsigned boff, int t) {
#pragma unroll
  for (int i = 0; i < 2; ++i) {
    int chunk = t + i * 256;           // 0..511
    int row   = chunk >> 2;
    int c8    = (chunk & 3) << 3;
    async_copy16(aoff + (unsigned)(row * LDSTR + c8) * 2,
                 (u64)(uintptr_t)(A + (size_t)(m0 + row) * K + k0 + c8));
  }
#pragma unroll
  for (int i = 0; i < 2; ++i) {
    int chunk = t + i * 256;
    int row   = chunk >> 2;
    int c8    = (chunk & 3) << 3;
    int n     = n0 + row;
    n = (n < N) ? n : (N - 1);         // clamp: junk cols never stored
    async_copy16(boff + (unsigned)(row * LDSTR + c8) * 2,
                 (u64)(uintptr_t)(Bm + (size_t)n * K + k0 + c8));
  }
}

// ------------------------- GEMM mainloop -----------------------------------
// C(MxN) = A(MxK) @ B(NxK)^T ; 128x128 block, BK=32, 8 waves of 32x64,
// double-buffered LDS fed by async DMA.
__device__ inline void gemm_mainloop(const __bf16* __restrict__ A,
                                     const __bf16* __restrict__ Bm,
                                     int N, int K, int m0, int n0,
                                     __bf16* ldsA, __bf16* ldsB, v8f acc[2][4]) {
  const int t    = threadIdx.x;
  const int lane = t & 31;
  const int w    = t >> 5;
  const int wm   = w >> 1;   // 0..3
  const int wn   = w & 1;    // 0..1

  v8f zero = {0.f, 0.f, 0.f, 0.f, 0.f, 0.f, 0.f, 0.f};
  for (int rt = 0; rt < 2; ++rt)
    for (int ct = 0; ct < 4; ++ct) acc[rt][ct] = zero;

  unsigned aoff[2] = { lds_addr32(ldsA), lds_addr32(ldsA + 128 * LDSTR) };
  unsigned boff[2] = { lds_addr32(ldsB), lds_addr32(ldsB + 128 * LDSTR) };

  const int nsteps = K >> 5;
  gemm_issue(A, Bm, N, K, m0, n0, 0, aoff[0], boff[0], t);

  for (int s = 0; s < nsteps; ++s) {
    int cur = s & 1;
    if (s + 1 < nsteps) {
      gemm_issue(A, Bm, N, K, m0, n0, (s + 1) << 5, aoff[cur ^ 1], boff[cur ^ 1], t);
      WAIT_ASYNC_4;                    // current buffer's 4 copies complete
    } else {
      WAIT_ASYNC_0;
    }
    __syncthreads();

    const __bf16* la = ldsA + cur * 128 * LDSTR;
    const __bf16* lb = ldsB + cur * 128 * LDSTR;
    v16bf af[2], bfr[4];
    for (int rt = 0; rt < 2; ++rt) af[rt]  = frag_a_lds(la, wm * 32 + rt * 16, lane);
    for (int ct = 0; ct < 4; ++ct) bfr[ct] = frag_b_lds(lb, wn * 64 + ct * 16, lane);
    for (int rt = 0; rt < 2; ++rt)
      for (int ct = 0; ct < 4; ++ct)
        acc[rt][ct] = WMMA_BF16(af[rt], bfr[ct], acc[rt][ct]);

    __syncthreads();                   // everyone done reading before reuse
  }
}

// ------------------------- GEMM kernels ------------------------------------
__global__ __launch_bounds__(256) void gemm_bf16_kernel(const __bf16* __restrict__ A,
                                                        const __bf16* __restrict__ Bm,
                                                        __bf16* __restrict__ C,
                                                        int N, int K) {
  __shared__ __bf16 ldsA[2 * 128 * LDSTR];
  __shared__ __bf16 ldsB[2 * 128 * LDSTR];
  int m0 = blockIdx.x * 128, n0 = blockIdx.y * 128;
  v8f acc[2][4];
  gemm_mainloop(A, Bm, N, K, m0, n0, ldsA, ldsB, acc);
  int t = threadIdx.x, lane = t & 31, w = t >> 5, wm = w >> 1, wn = w & 1;
  for (int rt = 0; rt < 2; ++rt)
    for (int ct = 0; ct < 4; ++ct)
      for (int r = 0; r < 8; ++r) {
        int row = m0 + wm * 32 + rt * 16 + r + ((lane < 16) ? 0 : 8);
        int col = n0 + wn * 64 + ct * 16 + (lane & 15);
        if (col < N) C[(size_t)row * N + col] = (__bf16)acc[rt][ct][r];
      }
}

__global__ __launch_bounds__(256) void gemm_f32_kernel(const __bf16* __restrict__ A,
                                                       const __bf16* __restrict__ Bm,
                                                       float* __restrict__ C,
                                                       int N, int K) {
  __shared__ __bf16 ldsA[2 * 128 * LDSTR];
  __shared__ __bf16 ldsB[2 * 128 * LDSTR];
  int m0 = blockIdx.x * 128, n0 = blockIdx.y * 128;
  v8f acc[2][4];
  gemm_mainloop(A, Bm, N, K, m0, n0, ldsA, ldsB, acc);
  int t = threadIdx.x, lane = t & 31, w = t >> 5, wm = w >> 1, wn = w & 1;
  for (int rt = 0; rt < 2; ++rt)
    for (int ct = 0; ct < 4; ++ct)
      for (int r = 0; r < 8; ++r) {
        int row = m0 + wm * 32 + rt * 16 + r + ((lane < 16) ? 0 : 8);
        int col = n0 + wn * 64 + ct * 16 + (lane & 15);
        C[(size_t)row * N + col] = acc[rt][ct][r];
      }
}

// kvb GEMM: scatter epilogue -> k_nope into K(B,H,S,192) and V transposed (B,H,VD,S)
__global__ __launch_bounds__(256) void gemm_kvb_kernel(const __bf16* __restrict__ A,
                                                       const __bf16* __restrict__ Bm,
                                                       __bf16* __restrict__ kbuf,
                                                       __bf16* __restrict__ vtb,
                                                       int N, int K) {
  __shared__ __bf16 ldsA[2 * 128 * LDSTR];
  __shared__ __bf16 ldsB[2 * 128 * LDSTR];
  int m0 = blockIdx.x * 128, n0 = blockIdx.y * 128;
  v8f acc[2][4];
  gemm_mainloop(A, Bm, N, K, m0, n0, ldsA, ldsB, acc);
  int t = threadIdx.x, lane = t & 31, w = t >> 5, wm = w >> 1, wn = w & 1;
  for (int rt = 0; rt < 2; ++rt)
    for (int ct = 0; ct < 4; ++ct)
      for (int r = 0; r < 8; ++r) {
        int row = m0 + wm * 32 + rt * 16 + r + ((lane < 16) ? 0 : 8);   // token
        int e   = n0 + wn * 64 + ct * 16 + (lane & 15);                 // h*256 + j
        int b = row >> 11, s = row & (CS - 1);
        int h = e >> 8,   j = e & 255;
        float v = acc[rt][ct][r];
        if (j < 128)
          kbuf[(((size_t)(b * CH + h)) * CS + s) * CQK + j] = (__bf16)v;
        else
          vtb[(((size_t)(b * CH + h)) * CVD + (j - 128)) * CS + s] = (__bf16)v;
      }
}

// ------------------------- RoPE + RMSNorm glue -----------------------------
__global__ __launch_bounds__(256) void rope_norm_kernel(const __bf16* __restrict__ qkv,
                                                        const float* __restrict__ kvw,
                                                        const float* __restrict__ cosf,
                                                        const float* __restrict__ sinf,
                                                        __bf16* __restrict__ qbuf,
                                                        __bf16* __restrict__ kvcb,
                                                        __bf16* __restrict__ kbuf) {
  int tok = blockIdx.x;
  int b = tok >> 11, s = tok & (CS - 1);
  int t = threadIdx.x;
  const __bf16* row = qkv + (size_t)tok * CE1;

  // RMSNorm(kv_c)
  float v0 = (float)row[CH * CQK + t];
  float v1 = (float)row[CH * CQK + 256 + t];
  __shared__ float red[256];
  red[t] = v0 * v0 + v1 * v1;
  __syncthreads();
  for (int off = 128; off > 0; off >>= 1) {
    if (t < off) red[t] += red[t + off];
    __syncthreads();
  }
  __shared__ float rs_sh;
  if (t == 0) rs_sh = rsqrtf(red[0] / (float)CKVR + 1e-6f);
  __syncthreads();
  float rs = rs_sh;
  kvcb[(size_t)tok * CKVR + t]       = (__bf16)(v0 * rs * kvw[t]);
  kvcb[(size_t)tok * CKVR + 256 + t] = (__bf16)(v1 * rs * kvw[256 + t]);

  const float* cr = cosf + (size_t)s * 32;
  const float* sr = sinf + (size_t)s * 32;

  // q: 1536 element-pairs -> (B,H,S,192), rope on last 64 of each head
  for (int p = t; p < CH * CQK / 2; p += 256) {
    int e = 2 * p;
    int h = e / CQK, d = e % CQK;
    float xe = (float)row[e], xo = (float)row[e + 1];
    size_t dst = (((size_t)(b * CH + h)) * CS + s) * CQK + d;
    if (d < 128) {
      qbuf[dst]     = (__bf16)xe;
      qbuf[dst + 1] = (__bf16)xo;
    } else {
      int j = (d - 128) >> 1;
      float c = cr[j], sn = sr[j];
      qbuf[dst]     = (__bf16)(xe * c - xo * sn);
      qbuf[dst + 1] = (__bf16)(xe * sn + xo * c);
    }
  }

  // k_pe: rope then broadcast into all heads' K[...,128:192]
  if (t < 32) {
    int j = t;
    float xe = (float)row[CH * CQK + CKVR + 2 * j];
    float xo = (float)row[CH * CQK + CKVR + 2 * j + 1];
    float c = cr[j], sn = sr[j];
    __bf16 ye = (__bf16)(xe * c - xo * sn);
    __bf16 yo = (__bf16)(xe * sn + xo * c);
    for (int h = 0; h < CH; ++h) {
      size_t dst = (((size_t)(b * CH + h)) * CS + s) * CQK + 128 + 2 * j;
      kbuf[dst]     = ye;
      kbuf[dst + 1] = yo;
    }
  }
}

// ------------------------- attention tile DMA ------------------------------
// 5 async 16B copies per thread: K tile 32x192 (3) + Vt tile 128x32 (2).
__device__ inline void attn_issue(const __bf16* __restrict__ Kp,
                                  const __bf16* __restrict__ Vt,
                                  int kstart, unsigned koff, unsigned voff, int t) {
  for (int i = t; i < 768; i += 256) {
    int krow = i / 24;
    int c8   = (i % 24) * 8;
    async_copy16(koff + (unsigned)(krow * KSTR + c8) * 2,
                 (u64)(uintptr_t)(Kp + (size_t)(kstart + krow) * CQK + c8));
  }
  for (int i = t; i < 512; i += 256) {
    int vrow = i >> 2;
    int c8   = (i & 3) * 8;
    async_copy16(voff + (unsigned)(vrow * LDSTR + c8) * 2,
                 (u64)(uintptr_t)(Vt + (size_t)vrow * CS + kstart + c8));
  }
}

// ------------------------- Flash attention (causal) ------------------------
// grid (S/128, H, B); 8 waves, each owns 16 query rows. Key blocks of 32,
// double-buffered async K/Vt tiles.
__global__ __launch_bounds__(256) void attn_kernel(const __bf16* __restrict__ qbuf,
                                                   const __bf16* __restrict__ kbuf,
                                                   const __bf16* __restrict__ vtb,
                                                   __bf16* __restrict__ attnb) {
  __shared__ __bf16 kl[2 * 32 * KSTR];      // 2 x (32 keys x 192 padded)
  __shared__ __bf16 vl[2 * 128 * LDSTR];    // 2 x (Vt: 128 vd x 32 keys padded)
  __shared__ __bf16 pl[8 * 16 * LDSTR];     // per-wave P tile: 16 x 32 (padded)

  int qblk = blockIdx.x, h = blockIdx.y, b = blockIdx.z;
  int t = threadIdx.x, lane = t & 31, w = t >> 5;
  int q0 = qblk * 128 + w * 16;

  const __bf16* Q  = qbuf + (((size_t)(b * CH + h)) * CS) * CQK;
  const __bf16* Kp = kbuf + (((size_t)(b * CH + h)) * CS) * CQK;
  const __bf16* Vt = vtb  + (((size_t)(b * CH + h)) * CVD) * CS;

  // preload Q fragments (6 d-chunks of 32)
  v16bf qf[6];
  {
    int row = q0 + (lane & 15);
    int kb  = (lane < 16) ? 0 : 8;
    for (int c = 0; c < 6; ++c) {
      V16U u;
      u.h[0] = *(const v8bf*)(Q + (size_t)row * CQK + 32 * c + kb);
      u.h[1] = *(const v8bf*)(Q + (size_t)row * CQK + 32 * c + kb + 16);
      qf[c] = u.v;
    }
  }

  v8f zero = {0.f, 0.f, 0.f, 0.f, 0.f, 0.f, 0.f, 0.f};
  v8f o[8];
  float mrow[8], lrow[8];
  for (int i = 0; i < 8; ++i) { o[i] = zero; mrow[i] = -3.0e38f; lrow[i] = 0.f; }

  const float CEXP = 1.4426950408889634f * 0.07216878364870323f; // log2e * QK^-0.5
  __bf16* pw = pl + w * 16 * LDSTR;
  unsigned koff[2] = { lds_addr32(kl), lds_addr32(kl + 32 * KSTR) };
  unsigned voff[2] = { lds_addr32(vl), lds_addr32(vl + 128 * LDSTR) };
  const int nkb = qblk * 4 + 4;

  attn_issue(Kp, Vt, 0, koff[0], voff[0], t);

  for (int kb_i = 0; kb_i < nkb; ++kb_i) {
    int kstart = kb_i * 32;
    int cur    = kb_i & 1;
    if (kb_i + 1 < nkb) {
      attn_issue(Kp, Vt, kstart + 32, koff[cur ^ 1], voff[cur ^ 1], t);
      WAIT_ASYNC_5;
    } else {
      WAIT_ASYNC_0;
    }
    __syncthreads();

    if (kstart <= q0 + 15) {   // wave-uniform: EXEC all ones around WMMA
      const __bf16* klc = kl + cur * 32 * KSTR;
      const __bf16* vlc = vl + cur * 128 * LDSTR;

      // scores = Q @ K^T for 16 queries x 32 keys
      v8f sc[2];
      sc[0] = zero; sc[1] = zero;
      for (int kt = 0; kt < 2; ++kt)
        for (int c = 0; c < 6; ++c) {
          int col = kt * 16 + (lane & 15);
          int kb2 = (lane < 16) ? 0 : 8;
          V16U u;
          u.h[0] = *(const v8bf*)(klc + col * KSTR + 32 * c + kb2);
          u.h[1] = *(const v8bf*)(klc + col * KSTR + 32 * c + kb2 + 16);
          sc[kt] = WMMA_BF16(qf[c], u.v, sc[kt]);
        }

      // causal mask + online softmax (row stats live in C-layout halves)
      int rbase = q0 + ((lane < 16) ? 0 : 8);
      int col0  = kstart + (lane & 15);
      float p0a[8], p1a[8];
      for (int r = 0; r < 8; ++r) {
        int rowg = rbase + r;
        float s0 = sc[0][r]; if (col0 > rowg)      s0 = -3.0e38f;
        float s1 = sc[1][r]; if (col0 + 16 > rowg) s1 = -3.0e38f;
        float mx = fmaxf(s0, s1);
        for (int off = 8; off; off >>= 1) mx = fmaxf(mx, __shfl_xor(mx, off, 32));
        float mnew = fmaxf(mrow[r], mx);
        float corr = exp2f((mrow[r] - mnew) * CEXP);
        float p0 = exp2f((s0 - mnew) * CEXP);
        float p1 = exp2f((s1 - mnew) * CEXP);
        float rsum = p0 + p1;
        for (int off = 8; off; off >>= 1) rsum += __shfl_xor(rsum, off, 32);
        lrow[r] = lrow[r] * corr + rsum;
        mrow[r] = mnew;
        for (int nt = 0; nt < 8; ++nt) o[nt][r] *= corr;
        p0a[r] = p0; p1a[r] = p1;
      }

      // bounce P through per-wave LDS: C-layout -> A-layout (LDS in-order per wave)
      {
        int prow = (lane < 16) ? 0 : 8;
        int pcol = lane & 15;
        for (int r = 0; r < 8; ++r) {
          pw[(prow + r) * LDSTR + pcol]      = (__bf16)p0a[r];
          pw[(prow + r) * LDSTR + 16 + pcol] = (__bf16)p1a[r];
        }
      }
      v16bf pf;
      {
        int row = lane & 15;
        int kb  = (lane < 16) ? 0 : 8;
        V16U u;
        u.h[0] = *(const v8bf*)(pw + row * LDSTR + kb);
        u.h[1] = *(const v8bf*)(pw + row * LDSTR + kb + 16);
        pf = u.v;
      }
      // O += P @ V  (contraction over 32 keys; Vt gives contiguous K per lane)
      for (int nt = 0; nt < 8; ++nt) {
        int vdrow = nt * 16 + (lane & 15);
        int kb2   = (lane < 16) ? 0 : 8;
        V16U u;
        u.h[0] = *(const v8bf*)(vlc + vdrow * LDSTR + kb2);
        u.h[1] = *(const v8bf*)(vlc + vdrow * LDSTR + kb2 + 16);
        o[nt] = WMMA_BF16(pf, u.v, o[nt]);
      }
    }
    __syncthreads();
  }

  // normalize and write (B, S, H*VD) bf16
  for (int r = 0; r < 8; ++r) {
    int rowg  = q0 + r + ((lane < 16) ? 0 : 8);
    float inv = 1.f / lrow[r];
    for (int nt = 0; nt < 8; ++nt) {
      int col = nt * 16 + (lane & 15);
      attnb[((size_t)(b * CS + rowg)) * (CH * CVD) + h * CVD + col] =
          (__bf16)(o[nt][r] * inv);
    }
  }
}

// ---------------------------------------------------------------------------
extern "C" void kernel_launch(void* const* d_in, const int* in_sizes, int n_in,
                              void* d_out, int out_size, void* d_ws, size_t ws_size,
                              hipStream_t stream) {
  (void)in_sizes; (void)n_in; (void)out_size; (void)ws_size;
  const float* x    = (const float*)d_in[0];
  const float* wq   = (const float*)d_in[1];
  const float* wkva = (const float*)d_in[2];
  const float* kvnw = (const float*)d_in[3];
  const float* wkvb = (const float*)d_in[4];
  const float* wo   = (const float*)d_in[5];
  const float* fcos = (const float*)d_in[6];
  const float* fsin = (const float*)d_in[7];
  float* out = (float*)d_out;

  char* base = (char*)d_ws;
  size_t off = 0;
  auto alloc = [&](size_t elems) {
    void* p = base + off;
    off += (elems * sizeof(__bf16) + 255) & ~(size_t)255;
    return (__bf16*)p;
  };
  __bf16* xb    = alloc((size_t)CM * CD);              // x bf16
  __bf16* w1b   = alloc((size_t)CE1 * CD);             // [wq ; wkv_a] bf16
  __bf16* wkvbb = alloc((size_t)4096 * CKVR);          // wkv_b bf16
  __bf16* wob   = alloc((size_t)CD * CD);              // wo bf16
  __bf16* qkvb  = alloc((size_t)CM * CE1);             // GEMM1 out
  __bf16* qbuf  = alloc((size_t)CB * CH * CS * CQK);   // Q (roped)
  __bf16* kvcb  = alloc((size_t)CM * CKVR);            // normed kv_c
  __bf16* kbuf  = alloc((size_t)CB * CH * CS * CQK);   // K (nope|pe)
  __bf16* vtb   = alloc((size_t)CB * CH * CVD * CS);   // V transposed
  __bf16* attnb = alloc((size_t)CM * CH * CVD);        // attention out

  cvt_kernel<<<2048, 256, 0, stream>>>(x,    xb,    CM * CD);
  cvt_kernel<<<2048, 256, 0, stream>>>(wq,   w1b,   CH * CQK * CD);
  cvt_kernel<<<2048, 256, 0, stream>>>(wkva, w1b + (size_t)CH * CQK * CD, (CKVR + 64) * CD);
  cvt_kernel<<<2048, 256, 0, stream>>>(wkvb, wkvbb, 4096 * CKVR);
  cvt_kernel<<<2048, 256, 0, stream>>>(wo,   wob,   CD * CD);

  gemm_bf16_kernel<<<dim3(CM / 128, (CE1 + 127) / 128), 256, 0, stream>>>(
      xb, w1b, qkvb, CE1, CD);
  rope_norm_kernel<<<CM, 256, 0, stream>>>(qkvb, kvnw, fcos, fsin, qbuf, kvcb, kbuf);
  gemm_kvb_kernel<<<dim3(CM / 128, 4096 / 128), 256, 0, stream>>>(
      kvcb, wkvbb, kbuf, vtb, 4096, CKVR);
  attn_kernel<<<dim3(CS / 128, CH, CB), 256, 0, stream>>>(qbuf, kbuf, vtb, attnb);
  gemm_f32_kernel<<<dim3(CM / 128, CD / 128), 256, 0, stream>>>(
      attnb, wob, out, CD, CD);
}
